// SelfAttention_78503412236928
// MI455X (gfx1250) — compile-verified, matrix operand order checked by reading, not run on hardware
//
#include <hip/hip_runtime.h>
#include <hip/hip_bf16.h>

typedef __attribute__((ext_vector_type(16))) _Float16 v16h;
typedef __attribute__((ext_vector_type(8)))  _Float16 v8h;
typedef __attribute__((ext_vector_type(4)))  _Float16 v4h;
typedef __attribute__((ext_vector_type(8)))  float    v8f;
typedef __attribute__((ext_vector_type(4)))  int      v4i;

#define DEV __device__ __forceinline__

#define NNq  4096   // pixels per batch
#define CAd  64
#define CHd  256
#define KCk  32     // key chunk in attention

#if __has_builtin(__builtin_amdgcn_global_load_async_to_lds_b128)
#define USE_ASYNC 1
typedef __attribute__((address_space(1))) v4i GV4i;  // global
typedef __attribute__((address_space(3))) v4i LV4i;  // LDS
#endif

// D = A(16x32 f16) x B(32x16 f16) + C(16x16 f32)
DEV v8f wmma_f16(v16h a, v16h b, v8f c) {
  return __builtin_amdgcn_wmma_f32_16x16x32_f16(
      false, a, false, b, (short)0, c, false, false);
}

// 16x32 f16 fragment from row-major [rows x ld] f16 buffer.
// lane: row = rbase + (lane&15); kb = (lane&16)?8:0
// halfs 0..7 <-> K k0+kb..+7 ; halfs 8..15 <-> K k0+kb+16..+23
DEV v16h load_frag(const _Float16* base, int rbase, int ld, int k0, int lane) {
  int kb = (lane & 16) ? 8 : 0;
  const _Float16* p = base + (long)(rbase + (lane & 15)) * ld + k0 + kb;
  v8h lo = *(const v8h*)(p);
  v8h hi = *(const v8h*)(p + 16);
  v16h r;
#pragma unroll
  for (int i = 0; i < 8; ++i) { r[i] = lo[i]; r[i + 8] = hi[i]; }
  return r;
}

// 16-byte global -> LDS copy: async on gfx1250.
DEV void copy16(const _Float16* g, _Float16* l) {
#ifdef USE_ASYNC
  __builtin_amdgcn_global_load_async_to_lds_b128((GV4i*)g, (LV4i*)l, 0, 0);
#else
  *(v8h*)l = *(const v8h*)g;
#endif
}

DEV void wait_async_all() {
#ifdef USE_ASYNC
#if __has_builtin(__builtin_amdgcn_s_wait_asynccnt)
  __builtin_amdgcn_s_wait_asynccnt(0);
#else
  asm volatile("s_wait_asynccnt 0x0" ::: "memory");
#endif
#endif
}

// ---------------------------------------------------------------------------
// Pre-pass: x (f32) -> f16, vectorized
// ---------------------------------------------------------------------------
__global__ __launch_bounds__(256)
void sa_cvt_x(const float* __restrict__ s, _Float16* __restrict__ d, int n4) {
  int i = blockIdx.x * 256 + threadIdx.x;
  if (i >= n4) return;
  float4 v = ((const float4*)s)[i];
  v4h o;
  o[0] = (_Float16)v.x; o[1] = (_Float16)v.y;
  o[2] = (_Float16)v.z; o[3] = (_Float16)v.w;
  ((v4h*)d)[i] = o;
}

// Pre-pass: W [K x NCOL] f32 -> Wt [NCOL x K] f16 (K = 256)
template <int NCOL>
__global__ __launch_bounds__(256)
void sa_cvt_wt(const float* __restrict__ W, _Float16* __restrict__ Wt) {
  int idx = blockIdx.x * 256 + threadIdx.x;   // over 256*NCOL
  int k = idx / NCOL;                          // NCOL pow2 -> shift
  int n = idx - k * NCOL;
  Wt[n * 256 + k] = (_Float16)W[idx];
}

// ---------------------------------------------------------------------------
// Projection GEMM: out(f16) = A(f16)[M,K] * Wt(f16)[N,K]^T + bias
// One wave per 16x16 tile. TRANSPOSED writes out[b][col][n] (n within batch).
// ---------------------------------------------------------------------------
template <int NTN, bool TRANSPOSED>
__global__ __launch_bounds__(256)
void sa_gemm(const _Float16* __restrict__ A, const _Float16* __restrict__ Wt,
             const float* __restrict__ bias, _Float16* __restrict__ out,
             int M, int K) {
  const int N = NTN * 16;
  int lane = threadIdx.x & 31;
  int wave = threadIdx.x >> 5;
  int tile = blockIdx.x * 8 + wave;
  int mt = tile / NTN;                // power-of-two: shift
  int nt = tile - mt * NTN;
  if (mt * 16 >= M) return;
  int r0 = mt * 16, c0 = nt * 16;
  int bcol = c0 + (lane & 15);

  v8f acc;
#pragma unroll
  for (int j = 0; j < 8; ++j) acc[j] = 0.0f;

#pragma unroll 4
  for (int k0 = 0; k0 < K; k0 += 32) {
    v16h af = load_frag(A, r0, K, k0, lane);
    v16h bf = load_frag(Wt, c0, K, k0, lane);
    acc = wmma_f16(af, bf, acc);
  }

  float bv = bias[bcol];
  int half8 = (lane & 16) ? 8 : 0;
#pragma unroll
  for (int j = 0; j < 8; ++j) {
    int gr = r0 + j + half8;
    float v = acc[j] + bv;
    long idx;
    if (TRANSPOSED) {
      int batch = gr >> 12;           // / 4096
      int n = gr & 4095;
      idx = (long)batch * N * NNq + (long)bcol * NNq + n;
    } else {
      idx = (long)gr * N + bcol;
    }
    out[idx] = (_Float16)v;
  }
}

// ---------------------------------------------------------------------------
// Flash-attention core. G,F: [B*N,64] f16; HT: [B][256][N] f16.
// 8 waves/block, 64 queries/block. Waves pair up on 16 queries; each wave
// owns 128 of the 256 channels (o[8] accumulators -> low VGPR pressure).
// 32-key chunks double-buffered in LDS with async global->LDS copies.
// ---------------------------------------------------------------------------
__global__ __launch_bounds__(256)
void sa_attn(const _Float16* __restrict__ G, const _Float16* __restrict__ F,
             const _Float16* __restrict__ HT, _Float16* __restrict__ O) {
  __shared__ _Float16 sF[2 * KCk * CAd];   // 2 x 4 KB
  __shared__ _Float16 sHT[2 * CHd * KCk];  // 2 x 16 KB

  int lane  = threadIdx.x & 31;
  int wave  = threadIdx.x >> 5;
  int qg    = wave >> 1;                   // 4 query groups of 16
  int chh   = (wave & 1) * 8;              // channel-tile half: tiles chh..chh+7
  int b     = blockIdx.x >> 6;             // 64 query-blocks per batch
  int qblk  = blockIdx.x & 63;
  int q0    = qblk * 64 + qg * 16;
  int qrow  = b * NNq + q0;
  int half8 = (lane & 16) ? 8 : 0;
  int tid   = threadIdx.x;

  const _Float16* Fb = F + (long)b * NNq * CAd;
  const _Float16* Hb = HT + (long)b * CHd * NNq;

  // Q fragments (also serve as B-operand of St = F x Q^T)
  v16h qf0 = load_frag(G, qrow, CAd, 0, lane);
  v16h qf1 = load_frag(G, qrow, CAd, 32, lane);

  v8f o[8];
#pragma unroll
  for (int t = 0; t < 8; ++t)
#pragma unroll
    for (int j = 0; j < 8; ++j) o[t][j] = 0.0f;

  float m = -3.0e38f, l = 0.0f;

  // stage one 32-key chunk: F tile 32x64 (contiguous), HT tile 256x32
  auto stage = [&](int buf, int k0) {
    _Float16* dF = sF + buf * (KCk * CAd);
    _Float16* dH = sHT + buf * (CHd * KCk);
    const _Float16* gF = Fb + (long)k0 * CAd;   // 2048 contiguous halfs
    copy16(gF + tid * 8, dF + tid * 8);         // 256 segments of 8 halfs
    const _Float16* gH = Hb + k0;
#pragma unroll
    for (int s = 0; s < 4; ++s) {
      int seg = tid + s * 256;                  // 1024 segments of 8 halfs
      int ch = seg >> 2;
      int off = (seg & 3) * 8;
      copy16(gH + (long)ch * NNq + off, dH + ch * KCk + off);
    }
  };

  stage(0, 0);
  wait_async_all();
  __syncthreads();

  for (int k0 = 0; k0 < NNq; k0 += KCk) {
    int buf = (k0 >> 5) & 1;
    if (k0 + KCk < NNq) stage(buf ^ 1, k0 + KCk);  // overlap with compute

    const _Float16* cF = sF + buf * (KCk * CAd);
    const _Float16* cH = sHT + buf * (CHd * KCk);

    // St (transposed scores) [16 key x 16 q] tiles, K = CA = 64
    v16h a00 = load_frag(cF, 0,  CAd, 0,  lane);
    v16h a01 = load_frag(cF, 0,  CAd, 32, lane);
    v16h a10 = load_frag(cF, 16, CAd, 0,  lane);
    v16h a11 = load_frag(cF, 16, CAd, 32, lane);
    v8f st0, st1;
#pragma unroll
    for (int j = 0; j < 8; ++j) { st0[j] = 0.0f; st1[j] = 0.0f; }
    st0 = wmma_f16(a00, qf0, st0);
    st0 = wmma_f16(a01, qf1, st0);
    st1 = wmma_f16(a10, qf0, st1);
    st1 = wmma_f16(a11, qf1, st1);

    // online softmax; per query = lane&15, halves merged with shfl_xor(16)
    float cm = -3.0e38f;
#pragma unroll
    for (int j = 0; j < 8; ++j) cm = fmaxf(cm, fmaxf(st0[j], st1[j]));
    cm = fmaxf(cm, __shfl_xor(cm, 16, 32));
    float mn = fmaxf(m, cm);
    float alpha = __expf(m - mn);
    m = mn;

    float p0[8], p1[8], cs = 0.0f;
#pragma unroll
    for (int j = 0; j < 8; ++j) {
      p0[j] = __expf(st0[j] - mn);
      p1[j] = __expf(st1[j] - mn);
      cs += p0[j] + p1[j];
    }
    cs += __shfl_xor(cs, 16, 32);
    l = l * alpha + cs;

    // P fragment: St layout == A-layout for O += P @ H
    v16h pf;
#pragma unroll
    for (int j = 0; j < 8; ++j) {
      pf[j]     = (_Float16)p0[j];
      pf[j + 8] = (_Float16)p1[j];
    }

    // rescale O by per-row alpha (8 bpermutes)
    float ar[8];
#pragma unroll
    for (int j = 0; j < 8; ++j) ar[j] = __shfl(alpha, j + half8, 32);
#pragma unroll
    for (int t = 0; t < 8; ++t)
#pragma unroll
      for (int j = 0; j < 8; ++j) o[t][j] *= ar[j];

    // O += P[16x32] @ H[32x128] : this wave's 8 channel tiles
#pragma unroll
    for (int t = 0; t < 8; ++t) {
      v16h hb = load_frag(cH, (chh + t) * 16, KCk, 0, lane);
      o[t] = wmma_f16(pf, hb, o[t]);
    }

    wait_async_all();   // next buffer in place
    __syncthreads();    // everyone done reading current buffer
  }

  // normalize, store O f16 [B*N, 256]
  float lr[8];
#pragma unroll
  for (int j = 0; j < 8; ++j) {
    float lv = __shfl(l, j + half8, 32);
    lr[j] = lv > 0.0f ? 1.0f / lv : 0.0f;
  }
  int col = lane & 15;
#pragma unroll
  for (int t = 0; t < 8; ++t)
#pragma unroll
    for (int j = 0; j < 8; ++j) {
      long gr = (long)(b * NNq + q0 + j + half8);
      O[gr * CHd + (chh + t) * 16 + col] = (_Float16)(o[t][j] * lr[j]);
    }
}

// ---------------------------------------------------------------------------
// Output GEMM: y = gamma * (O @ Wo + bo) + x  (fp32 out, fused residual)
// ---------------------------------------------------------------------------
template <int NTN>
__global__ __launch_bounds__(256)
void sa_out(const _Float16* __restrict__ A, const _Float16* __restrict__ Wt,
            const float* __restrict__ bias, const float* __restrict__ x,
            const float* __restrict__ gamma, float* __restrict__ y,
            int M, int K) {
  const int N = NTN * 16;
  int lane = threadIdx.x & 31;
  int wave = threadIdx.x >> 5;
  int tile = blockIdx.x * 8 + wave;
  int mt = tile / NTN;
  int nt = tile - mt * NTN;
  if (mt * 16 >= M) return;
  int r0 = mt * 16, c0 = nt * 16;
  int bcol = c0 + (lane & 15);

  v8f acc;
#pragma unroll
  for (int j = 0; j < 8; ++j) acc[j] = 0.0f;

#pragma unroll 4
  for (int k0 = 0; k0 < K; k0 += 32) {
    v16h af = load_frag(A, r0, K, k0, lane);
    v16h bf = load_frag(Wt, c0, K, k0, lane);
    acc = wmma_f16(af, bf, acc);
  }

  float g = gamma[0];
  float bv = bias[bcol];
  int half8 = (lane & 16) ? 8 : 0;
#pragma unroll
  for (int j = 0; j < 8; ++j) {
    long idx = (long)(r0 + j + half8) * N + bcol;
    y[idx] = g * (acc[j] + bv) + x[idx];
  }
}

// ---------------------------------------------------------------------------
extern "C" void kernel_launch(void* const* d_in, const int* in_sizes, int n_in,
                              void* d_out, int out_size, void* d_ws, size_t ws_size,
                              hipStream_t stream) {
  (void)in_sizes; (void)n_in; (void)out_size; (void)ws_size;
  const float* x     = (const float*)d_in[0];
  const float* Wf    = (const float*)d_in[1];
  const float* bf    = (const float*)d_in[2];
  const float* Wg    = (const float*)d_in[3];
  const float* bg    = (const float*)d_in[4];
  const float* Wh    = (const float*)d_in[5];
  const float* bh    = (const float*)d_in[6];
  const float* Wo    = (const float*)d_in[7];
  const float* bo    = (const float*)d_in[8];
  const float* gamma = (const float*)d_in[9];

  const int Bb = 4, C = 256, CA = 64, CH = 256;
  const int M = Bb * NNq;  // 16384

  char* ws = (char*)d_ws;
  size_t off = 0;
  _Float16* xh    = (_Float16*)(ws + off); off += (size_t)M * C * 2;    // 8 MB
  _Float16* Fbuf  = (_Float16*)(ws + off); off += (size_t)M * CA * 2;   // 2 MB
  _Float16* Gbuf  = (_Float16*)(ws + off); off += (size_t)M * CA * 2;   // 2 MB
  _Float16* HTbuf = (_Float16*)(ws + off); off += (size_t)M * CH * 2;   // 8 MB
  _Float16* Obuf  = (_Float16*)(ws + off); off += (size_t)M * CH * 2;   // 8 MB
  _Float16* WfT   = (_Float16*)(ws + off); off += (size_t)C * CA * 2;
  _Float16* WgT   = (_Float16*)(ws + off); off += (size_t)C * CA * 2;
  _Float16* WhT   = (_Float16*)(ws + off); off += (size_t)C * CH * 2;
  _Float16* WoT   = (_Float16*)(ws + off); off += (size_t)CH * CH * 2;

  // pre-passes: convert x to f16; convert+transpose weights to f16 [N][K]
  int n4 = M * C / 4;
  sa_cvt_x<<<(n4 + 255) / 256, 256, 0, stream>>>(x, xh, n4);
  sa_cvt_wt<64><<<(C * CA) / 256, 256, 0, stream>>>(Wf, WfT);
  sa_cvt_wt<64><<<(C * CA) / 256, 256, 0, stream>>>(Wg, WgT);
  sa_cvt_wt<256><<<(C * CH) / 256, 256, 0, stream>>>(Wh, WhT);
  sa_cvt_wt<256><<<(CH * CH) / 256, 256, 0, stream>>>(Wo, WoT);

  // projections: f, g row-major; h transposed per batch [b][ch][n]
  int tilesFG = (M / 16) * (CA / 16);   // 4096
  sa_gemm<4, false><<<tilesFG / 8, 256, 0, stream>>>(xh, WfT, bf, Fbuf, M, C);
  sa_gemm<4, false><<<tilesFG / 8, 256, 0, stream>>>(xh, WgT, bg, Gbuf, M, C);
  int tilesH = (M / 16) * (CH / 16);    // 16384
  sa_gemm<16, true><<<tilesH / 8, 256, 0, stream>>>(xh, WhT, bh, HTbuf, M, C);

  // flash attention: 4 batches x 64 query-blocks, 8 waves / 64 queries each
  sa_attn<<<Bb * (NNq / 64), 256, 0, stream>>>(Gbuf, Fbuf, HTbuf, Obuf);

  // output projection + residual
  int tilesO = (M / 16) * (CH / 16);
  sa_out<16><<<tilesO / 8, 256, 0, stream>>>(Obuf, WoT, bo, x, gamma,
                                             (float*)d_out, M, CH);
}
